// NonLocalEmbeddedGaussianBlock2D_15358803050801
// MI455X (gfx1250) — matrix operand order repaired and checked by reading, not compiled, not run on hardware
//
#include <hip/hip_runtime.h>

#define B_  8
#define C_  128
#define N_  4096
#define IC_ 64

typedef _Float16 v16h __attribute__((ext_vector_type(16)));
typedef _Float16 v8h  __attribute__((ext_vector_type(8)));
typedef float    v8f  __attribute__((ext_vector_type(8)));

static __device__ __forceinline__ v16h cat8(v8h a, v8h b) {
  v16h r;
#pragma unroll
  for (int i = 0; i < 8; ++i) { r[i] = a[i]; r[i + 8] = b[i]; }
  return r;
}

static __device__ __forceinline__ v8f wmma_f16(v16h a, v16h b, v8f c) {
  // D = A(16x32 f16) * B(32x16 f16) + C(16x16 f32)
  return __builtin_amdgcn_wmma_f32_16x16x32_f16(false, a, false, b, (short)0, c,
                                                false, false);
}

// gfx1250 async Global -> LDS copy (ASYNCcnt-tracked), 16 bytes per lane.
static __device__ __forceinline__ void async_load_b128(const void* gptr, void* lptr) {
  unsigned lds_off = (unsigned)(uintptr_t)lptr;  // low 32 bits = LDS byte offset
  asm volatile("global_load_async_to_lds_b128 %0, %1, off"
               :
               : "v"(lds_off), "v"(gptr)
               : "memory");
}

// ---------------------------------------------------------------------------
// Kernel 1: convert weights to f16, fold BN into per-channel scale/shift.
// ---------------------------------------------------------------------------
__global__ void prep_kernel(const float* __restrict__ Wg, const float* __restrict__ Wt,
                            const float* __restrict__ Wp, const float* __restrict__ Wo,
                            const float* __restrict__ gamma, const float* __restrict__ beta,
                            const float* __restrict__ mean, const float* __restrict__ var,
                            _Float16* __restrict__ WgH, _Float16* __restrict__ WtH,
                            _Float16* __restrict__ WpH, _Float16* __restrict__ WoH,
                            float* __restrict__ bnS, float* __restrict__ bnB) {
  int i = blockIdx.x * blockDim.x + threadIdx.x;
  if (i < IC_ * C_) {           // all four weight matrices have 8192 elements
    WgH[i] = (_Float16)Wg[i];
    WtH[i] = (_Float16)Wt[i];
    WpH[i] = (_Float16)Wp[i];
    WoH[i] = (_Float16)Wo[i];
  }
  if (i < C_) {
    float s = gamma[i] * rsqrtf(var[i] + 1e-5f);
    bnS[i] = s;
    bnB[i] = beta[i] - mean[i] * s;
  }
}

// ---------------------------------------------------------------------------
// Kernel 2: input 1x1 convs (three GEMMs) via WMMA.
//   theta, phi  -> token-major [B, N, IC]  (f16)
//   g           -> channel-major [B, IC, N] (f16)
// One wave per 16-token tile; 8 waves / block.
// ---------------------------------------------------------------------------
__global__ void __launch_bounds__(256) proj_in_kernel(
    const float* __restrict__ x,
    const _Float16* __restrict__ WgH, const _Float16* __restrict__ WtH,
    const _Float16* __restrict__ WpH,
    const float* __restrict__ bg, const float* __restrict__ bt,
    const float* __restrict__ bp,
    _Float16* __restrict__ gT, _Float16* __restrict__ thetaT,
    _Float16* __restrict__ phiT) {
  __shared__ __align__(32) _Float16 xs[8][16][C_];   // per-wave x^T tile (32 KB)
  const int w    = threadIdx.x >> 5;
  const int lane = threadIdx.x & 31;
  const int c16  = lane & 15;
  const int h    = lane >> 4;
  const int tile = blockIdx.x * 8 + w;               // 2048 tiles total
  const int b    = tile >> 8;                        // 256 tiles per batch
  const int n0   = (tile & 255) << 4;

  // Stage x[b, :, n0..n0+15] into LDS transposed as f16.  x is [B][C][N].
  const size_t xbase = ((size_t)b * C_) * N_ + n0 + c16;
#pragma unroll 4
  for (int c0 = 0; c0 < C_; c0 += 2) {
    int c = c0 + h;
    xs[w][c16][c] = (_Float16)x[xbase + (size_t)c * N_];
  }

  // A fragments: 16 tokens x K=128 channels, four K-chunks of 32.
  v16h A[4];
#pragma unroll
  for (int f = 0; f < 4; ++f) {
    v8h a0 = *(const v8h*)&xs[w][c16][32 * f + 16 * h];
    v8h a1 = *(const v8h*)&xs[w][c16][32 * f + 16 * h + 8];
    A[f] = cat8(a0, a1);
  }

  // ---- theta = Wt @ x  (token-major store) ----
#pragma unroll
  for (int j = 0; j < 4; ++j) {
    v8f acc = {};
#pragma unroll
    for (int f = 0; f < 4; ++f) {
      v16h bf = *(const v16h*)&WtH[(16 * j + c16) * C_ + 32 * f + 16 * h];
      acc = wmma_f16(A[f], bf, acc);
    }
    float bias = bt[16 * j + c16];
    size_t obase = ((size_t)b * N_ + n0) * IC_ + 16 * j + c16;
#pragma unroll
    for (int r = 0; r < 8; ++r)
      thetaT[obase + (size_t)(r + 8 * h) * IC_] = (_Float16)(acc[r] + bias);
  }

  // ---- phi = Wp @ x  (token-major store) ----
#pragma unroll
  for (int j = 0; j < 4; ++j) {
    v8f acc = {};
#pragma unroll
    for (int f = 0; f < 4; ++f) {
      v16h bf = *(const v16h*)&WpH[(16 * j + c16) * C_ + 32 * f + 16 * h];
      acc = wmma_f16(A[f], bf, acc);
    }
    float bias = bp[16 * j + c16];
    size_t obase = ((size_t)b * N_ + n0) * IC_ + 16 * j + c16;
#pragma unroll
    for (int r = 0; r < 8; ++r)
      phiT[obase + (size_t)(r + 8 * h) * IC_] = (_Float16)(acc[r] + bias);
  }

  // ---- g = Wg @ x  (channel-major [IC][N]; 8 consecutive tokens -> one b128) ----
#pragma unroll
  for (int j = 0; j < 4; ++j) {
    v8f acc = {};
#pragma unroll
    for (int f = 0; f < 4; ++f) {
      v16h bf = *(const v16h*)&WgH[(16 * j + c16) * C_ + 32 * f + 16 * h];
      acc = wmma_f16(A[f], bf, acc);
    }
    float bias = bg[16 * j + c16];
    v8h gp;
#pragma unroll
    for (int r = 0; r < 8; ++r) gp[r] = (_Float16)(acc[r] + bias);
    *(v8h*)&gT[((size_t)b * IC_ + 16 * j + c16) * N_ + n0 + 8 * h] = gp;
  }
}

// ---------------------------------------------------------------------------
// Kernel 3: flash-attention with async-LDS staged K/V.
// Block = 8 waves, all on the same batch, 128 consecutive query rows.
// Keys processed in 64-wide chunks, double-buffered via ASYNCcnt:
//   issue copies(c+1) -> s_wait_asynccnt 4 -> barrier -> 16 WMMAs -> barrier.
// ---------------------------------------------------------------------------
__global__ void __launch_bounds__(256) attn_kernel(
    const _Float16* __restrict__ thetaT, const _Float16* __restrict__ phiT,
    const _Float16* __restrict__ gT, _Float16* __restrict__ y) {
  __shared__ __align__(32) _Float16 phis[2][64][64];   // keys token-major (16 KB)
  __shared__ __align__(32) _Float16 gs[2][64][64];     // values ic-major (16 KB)
  __shared__ __align__(32) _Float16 plds[8][16 * 64];  // per-wave P buffer (16 KB)
  const int tid  = threadIdx.x;
  const int w    = tid >> 5;
  const int lane = tid & 31;
  const int c16  = lane & 15;
  const int h    = lane >> 4;
  const int tile = blockIdx.x * 8 + w;
  const int b    = tile >> 8;
  const int q0   = (tile & 255) << 4;
  const size_t tokB = (size_t)b * N_;
  const size_t bIC  = (size_t)b * IC_;

  // Issue async copies of key/value chunk c into LDS buffer `buf`.
  // phi chunk: 64 tokens x 64 ic  -> 4096 contiguous f16 (512 x 16B segments).
  // g   chunk: 64 ic rows, 128B per row, row stride N_*2 bytes.
  auto issue = [&](int c, int buf) {
    const int m0 = c * 64;
    const char* phiG = (const char*)&phiT[(tokB + m0) * IC_];
    const char* gG   = (const char*)&gT[bIC * N_ + m0];
    char* phiL = (char*)&phis[buf][0][0];
    char* gL   = (char*)&gs[buf][0][0];
#pragma unroll
    for (int i = 0; i < 2; ++i) {
      int s = tid + i * 256;                       // segment id 0..511
      async_load_b128(phiG + s * 16, phiL + s * 16);
      async_load_b128(gG + (s >> 3) * (N_ * 2) + (s & 7) * 16, gL + s * 16);
    }
  };

  // Q A-fragments: 16 queries x K=IC(64), two chunks of 32.
  v16h QA[2];
  {
    const _Float16* qp = &thetaT[(tokB + q0 + c16) * IC_];
#pragma unroll
    for (int f = 0; f < 2; ++f)
      QA[f] = cat8(*(const v8h*)&qp[32 * f + 16 * h],
                   *(const v8h*)&qp[32 * f + 16 * h + 8]);
  }

  v8f O[4] = {};
  float mrow[8], lrow[8];
#pragma unroll
  for (int r = 0; r < 8; ++r) { mrow[r] = -3.0e38f; lrow[r] = 0.f; }

  _Float16* pb = &plds[w][0];
  const int NC = N_ / 64;  // 64 key chunks

  issue(0, 0);
  for (int c = 0; c < NC; ++c) {
    const int buf = c & 1;
    if (c + 1 < NC) {
      issue(c + 1, buf ^ 1);
      asm volatile("s_wait_asynccnt 0x4" ::: "memory");  // chunk c's copies done
    } else {
      asm volatile("s_wait_asynccnt 0x0" ::: "memory");
    }
    __syncthreads();

    // S = Q * phi^T for 64 keys (four 16x16 tiles, K=64)
    v8f S[4] = {};
#pragma unroll
    for (int kt = 0; kt < 4; ++kt) {
#pragma unroll
      for (int f = 0; f < 2; ++f) {
        v16h bf = *(const v16h*)&phis[buf][16 * kt + c16][32 * f + 16 * h];
        S[kt] = wmma_f16(QA[f], bf, S[kt]);
      }
    }

    // Online softmax over 64 keys. C-layout: element r = row (r+8h), col c16;
    // row reductions are xor-shuffles within each 16-lane half.
#pragma unroll
    for (int r = 0; r < 8; ++r) {
      float t = fmaxf(fmaxf(S[0][r], S[1][r]), fmaxf(S[2][r], S[3][r]));
      t = fmaxf(t, __shfl_xor(t, 1));
      t = fmaxf(t, __shfl_xor(t, 2));
      t = fmaxf(t, __shfl_xor(t, 4));
      t = fmaxf(t, __shfl_xor(t, 8));
      float nm = fmaxf(mrow[r], t);
      float sc = __expf(mrow[r] - nm);
      mrow[r] = nm;
      float rs = 0.f;
#pragma unroll
      for (int kt = 0; kt < 4; ++kt) {
        float p = __expf(S[kt][r] - nm);
        S[kt][r] = p;
        rs += p;
      }
      rs += __shfl_xor(rs, 1);
      rs += __shfl_xor(rs, 2);
      rs += __shfl_xor(rs, 4);
      rs += __shfl_xor(rs, 8);
      lrow[r] = lrow[r] * sc + rs;
#pragma unroll
      for (int j = 0; j < 4; ++j) O[j][r] *= sc;
    }

    // C-layout -> A-layout via wave-private LDS round trip (P is 16x64).
#pragma unroll
    for (int r = 0; r < 8; ++r) {
      int q = r + 8 * h;
#pragma unroll
      for (int kt = 0; kt < 4; ++kt)
        pb[q * 64 + 16 * kt + c16] = (_Float16)S[kt][r];
    }
    v16h PA0 = cat8(*(const v8h*)&pb[c16 * 64 + 16 * h],
                    *(const v8h*)&pb[c16 * 64 + 16 * h + 8]);
    v16h PA1 = cat8(*(const v8h*)&pb[c16 * 64 + 32 + 16 * h],
                    *(const v8h*)&pb[c16 * 64 + 40 + 16 * h]);

    // O += P(16x64) * g(64 keys x 64 ic); gs is ic-major so B-frags contiguous.
#pragma unroll
    for (int j = 0; j < 4; ++j) {
      const _Float16* gp = &gs[buf][16 * j + c16][0];
      O[j] = wmma_f16(PA0, *(const v16h*)&gp[16 * h], O[j]);
      O[j] = wmma_f16(PA1, *(const v16h*)&gp[32 + 16 * h], O[j]);
    }
    __syncthreads();  // done reading buf before it is overwritten two chunks on
  }

  // Normalize rows and store y token-major f16.
#pragma unroll
  for (int r = 0; r < 8; ++r) {
    float inv = 1.0f / lrow[r];
    size_t row = (tokB + q0 + r + 8 * h) * IC_ + c16;
#pragma unroll
    for (int j = 0; j < 4; ++j)
      y[row + 16 * j] = (_Float16)(O[j][r] * inv);
  }
}

// ---------------------------------------------------------------------------
// Kernel 4: output 1x1 conv (Wo @ y) + bias + folded BN + residual.
// ---------------------------------------------------------------------------
__global__ void __launch_bounds__(256) proj_out_kernel(
    const _Float16* __restrict__ y, const _Float16* __restrict__ WoH,
    const float* __restrict__ bo, const float* __restrict__ bnS,
    const float* __restrict__ bnB, const float* __restrict__ x,
    float* __restrict__ out) {
  const int w    = threadIdx.x >> 5;
  const int lane = threadIdx.x & 31;
  const int c16  = lane & 15;
  const int h    = lane >> 4;
  const int tile = blockIdx.x * 8 + w;
  const int b    = tile >> 8;
  const int n0   = (tile & 255) << 4;

  v16h YA[2];
  {
    const _Float16* yp = &y[((size_t)b * N_ + n0 + c16) * IC_];
#pragma unroll
    for (int f = 0; f < 2; ++f)
      YA[f] = cat8(*(const v8h*)&yp[32 * f + 16 * h],
                   *(const v8h*)&yp[32 * f + 16 * h + 8]);
  }

#pragma unroll
  for (int j = 0; j < 8; ++j) {
    v8f acc = {};
#pragma unroll
    for (int f = 0; f < 2; ++f) {
      v16h bf = *(const v16h*)&WoH[(16 * j + c16) * IC_ + 32 * f + 16 * h];
      acc = wmma_f16(YA[f], bf, acc);
    }
    int c = 16 * j + c16;
    float bias = bo[c], s = bnS[c], t = bnB[c];
    size_t base = ((size_t)b * C_ + c) * N_ + n0 + 8 * h;
#pragma unroll
    for (int r = 0; r < 8; ++r)
      out[base + r] = x[base + r] + (acc[r] + bias) * s + t;
  }
}

// ---------------------------------------------------------------------------
extern "C" void kernel_launch(void* const* d_in, const int* in_sizes, int n_in,
                              void* d_out, int out_size, void* d_ws, size_t ws_size,
                              hipStream_t stream) {
  (void)in_sizes; (void)n_in; (void)out_size; (void)ws_size;
  const float* x     = (const float*)d_in[0];
  const float* Wg    = (const float*)d_in[1];
  const float* bg    = (const float*)d_in[2];
  const float* Wt    = (const float*)d_in[3];
  const float* bt    = (const float*)d_in[4];
  const float* Wp    = (const float*)d_in[5];
  const float* bp    = (const float*)d_in[6];
  const float* Wo    = (const float*)d_in[7];
  const float* bo    = (const float*)d_in[8];
  const float* gamma = (const float*)d_in[9];
  const float* beta  = (const float*)d_in[10];
  const float* mean  = (const float*)d_in[11];
  const float* var   = (const float*)d_in[12];

  char* p = (char*)d_ws;
  const size_t tokBytes = (size_t)B_ * N_ * IC_ * sizeof(_Float16);  // 4 MB
  _Float16* thetaT = (_Float16*)p; p += tokBytes;
  _Float16* phiT   = (_Float16*)p; p += tokBytes;
  _Float16* gT     = (_Float16*)p; p += tokBytes;
  _Float16* yT     = (_Float16*)p; p += tokBytes;
  const size_t wBytes = (size_t)IC_ * C_ * sizeof(_Float16);         // 16 KB
  _Float16* WgH = (_Float16*)p; p += wBytes;
  _Float16* WtH = (_Float16*)p; p += wBytes;
  _Float16* WpH = (_Float16*)p; p += wBytes;
  _Float16* WoH = (_Float16*)p; p += wBytes;
  float* bnS = (float*)p; p += C_ * sizeof(float);
  float* bnB = (float*)p; p += C_ * sizeof(float);

  prep_kernel<<<(IC_ * C_ + 255) / 256, 256, 0, stream>>>(
      Wg, Wt, Wp, Wo, gamma, beta, mean, var, WgH, WtH, WpH, WoH, bnS, bnB);

  const int blocks = (B_ * (N_ / 16)) / 8;  // 256 blocks, 8 waves each
  proj_in_kernel<<<blocks, 256, 0, stream>>>(x, WgH, WtH, WpH, bg, bt, bp,
                                             gT, thetaT, phiT);
  attn_kernel<<<blocks, 256, 0, stream>>>(thetaT, phiT, gT, yT);
  proj_out_kernel<<<blocks, 256, 0, stream>>>(yT, WoH, bo, bnS, bnB, x,
                                              (float*)d_out);
}
